// ConditionalRandomField_5789615915448
// MI455X (gfx1250) — matrix operand order repaired and checked by reading, not compiled, not run on hardware
//
#include <hip/hip_runtime.h>
#include <hip/hip_bf16.h>
#include <stdint.h>

typedef __attribute__((ext_vector_type(16))) _Float16 v16h;
typedef __attribute__((ext_vector_type(8)))  _Float16 v8h;
typedef __attribute__((ext_vector_type(4)))  _Float16 v4h;
typedef __attribute__((ext_vector_type(8)))  float    v8f;
typedef __attribute__((ext_vector_type(4)))  unsigned int v4u;
typedef __attribute__((ext_vector_type(4)))  int      v4i;
typedef __attribute__((ext_vector_type(8)))  int      v8i;

#define BATCH   512
#define SEQ     1024
#define NTAGS   64
#define ROWS    16       // batch rows per workgroup
#define THREADS 128      // 4 waves (wave32)
#define EAS     72       // expA row stride (halves), 16B-aligned rows
#define ETS     72       // expT^T row stride (halves)
#define AS      68       // alpha row stride (floats), 16B-aligned rows
#define EMP     65       // TDM-padded emission row stride (64 + 1 pad DWORD)

#define LOG2E 1.4426950408889634f
#define LN2   0.6931471805599453f

// Dynamic-LDS layout (byte offsets; no static LDS => dynamic base is LDS addr 0,
// which the TDM descriptor's lds_addr field is relative to).
#define OFF_EXPTT 0                                  // 64*72*2  = 9216
#define OFF_EXPA  9216                               // 16*72*2  = 2304
#define OFF_ALPHA 11520                              // 16*68*4  = 4352
#define OFF_EMIT0 15872                              // 16*65*4  = 4160
#define OFF_EMIT1 20032                              // 16*65*4  = 4160
#define OFF_RM    24192                              // 16*8 = 128  ({rowmax, mask} pairs)
#define SMEM_BYTES 24320

// ---------------------------------------------------------------------------
// TDM: async-load one 16x64 f32 emission tile (row stride SEQ*64 elements in
// global) into LDS at lds_off, padding 1 DWORD per 64 DWORDs (row stride 65).
// D# per CDNA5 ISA. Tracked by TENSORcnt.
// ---------------------------------------------------------------------------
__device__ __forceinline__ void tdm_load_emit_tile(const float* gsrc, unsigned int lds_off)
{
    const uint64_t ga = (uint64_t)(uintptr_t)gsrc;
    // group0: count=1 | lds_addr | global_addr[56:0] | type=2
    v4u g0 = { 1u,
               lds_off,
               (unsigned int)(ga & 0xFFFFFFFFu),
               (unsigned int)(((ga >> 32) & 0x01FFFFFFu) | (2u << 30)) };
    // group1: data_size=4B(2), pad_enable=1, pad_interval=64DW(5), pad_amount=1DW(0)
    //         tensor_dim0=65536, tensor_dim1=512, tile_dim0=64, tile_dim1=16,
    //         tensor_dim0_stride=65536
    v8i g1 = { (int)0x01520000,
               0,
               (int)0x02000001,     // dim0[31:16]=1 (65536), dim1[15:0]=512
               (int)0x00400000,     // tile_dim0=64
               16,                  // tile_dim1=16, tile_dim2=0
               (int)0x00010000,     // tensor_dim0_stride=65536
               0, 0 };
    v4i g2 = { 1, 1, 0, 0 };        // tensor_dim2=1, tensor_dim3=1, tile_dim3=0
    v4i g3 = { 0, 0, 0, 0 };
#if defined(__clang_major__) && (__clang_major__ >= 23)
    v8i g4 = { 0, 0, 0, 0, 0, 0, 0, 0 };
    __builtin_amdgcn_tensor_load_to_lds(g0, g1, g2, g3, g4, 0);
#else
    __builtin_amdgcn_tensor_load_to_lds(g0, g1, g2, g3, 0);
#endif
}

// ---------------------------------------------------------------------------
// Forward algorithm (log-denominator) via exp-space GEMM on WMMA, entirely in
// base-2 log domain so exp/log lower to bare v_exp_f32/v_log_f32:
//   alpha2' = log2e*emit + m2 + log2( sum_i 2^(alpha2_i - m2) * 2^(T_ij*log2e) )
// Each lane keeps its owned 8 alpha elements in registers across the scan;
// LDS alpha is the published copy for the rowmax/expA broadcast passes.
// ---------------------------------------------------------------------------
__global__ __launch_bounds__(THREADS) void crf_forward_kernel(
    const float* __restrict__ inputs,
    const int*   __restrict__ mask,
    const float* __restrict__ transitions,
    const float* __restrict__ start_t,
    const float* __restrict__ end_t,
    float* __restrict__ denom_out)
{
    extern __shared__ char smem[];
    _Float16* expTt  = (_Float16*)(smem + OFF_EXPTT);   // expTt[j][i] = 2^(T[i][j]*log2e)
    _Float16* expA   = (_Float16*)(smem + OFF_EXPA);    // 2^(alpha2 - rowmax2)
    float*    alpha  = (float*)   (smem + OFF_ALPHA);   // log2-domain alpha (published)
    float*    emit0  = (float*)   (smem + OFF_EMIT0);   // TDM tile, row stride EMP
    float*    emit1  = (float*)   (smem + OFF_EMIT1);
    float2*   rm_s   = (float2*)  (smem + OFF_RM);      // {rowmax2, mask} per row

    const int tid  = threadIdx.x;
    const int lane = tid & 31;
    const int jb   = tid >> 5;              // wave id == output tag block (16 cols)
    const int b0   = blockIdx.x * ROWS;     // first batch row of this workgroup
    const float* gbase = inputs + (size_t)b0 * SEQ * NTAGS;

    // Kick off async emission tiles for t=0 and t=1 (double buffered).
    if (jb == 0) {
        tdm_load_emit_tile(gbase + 0 * NTAGS, OFF_EMIT0);
        tdm_load_emit_tile(gbase + 1 * NTAGS, OFF_EMIT1);
    }

    // 2^(T*log2e), transposed, into LDS (once)
    for (int e = tid; e < NTAGS * NTAGS; e += THREADS) {
        const int j = e >> 6, i = e & 63;
        expTt[j * ETS + i] = (_Float16)__builtin_amdgcn_exp2f(transitions[i * NTAGS + j] * LOG2E);
    }
    if (jb == 0) __builtin_amdgcn_s_wait_tensorcnt(1);   // t=0 tile landed
    __syncthreads();

    // Ownership map (constant all scan):
    //  A-layout: lane row mrow = lane&15, K-offset kbase = 16*(lane>>4)
    //  D-layout: lane column j = jb*16 + (lane&15); VGPR r -> row M = r + 8*(lane>>4)
    const int nloc  = lane & 15;
    const int kbase = (lane >> 4) * 16;
    const int mrow  = lane & 15;
    const int koff  = (lane >> 4) * 8;
    const int half8 = (lane >> 4) * 8;
    const int j     = jb * 16 + nloc;

    // alpha2_0 = (start + emit[0]) * log2e  (owner lanes, kept in registers)
    float alpha_reg[8];
    {
        const float sj = start_t[j];
        #pragma unroll
        for (int r = 0; r < 8; ++r) {
            const int M = r + half8;
            alpha_reg[r] = (sj + emit0[M * EMP + j]) * LOG2E;
            alpha[M * AS + j] = alpha_reg[r];
        }
    }

    // Hoist B fragments (time-invariant). B k order used here (same as A): lane L holds column
    // N = L&15, contiguous K starting at 16*(L>>4) within each K=32 chunk.
    v16h bfrag0, bfrag1;
    {
        const _Float16* brow = expTt + (jb * 16 + nloc) * ETS;
        ((v8h*)&bfrag0)[0] = *(const v8h*)(brow + 0  + kbase);
        ((v8h*)&bfrag0)[1] = *(const v8h*)(brow + 0  + kbase + 8);
        ((v8h*)&bfrag1)[0] = *(const v8h*)(brow + 32 + kbase);
        ((v8h*)&bfrag1)[1] = *(const v8h*)(brow + 32 + kbase + 8);
    }
    __syncthreads();

    for (int t = 1; t < SEQ; ++t) {
        const float* emit_s = (t & 1) ? emit1 : emit0;

        // (a0) prefetch t+1 tile into the other buffer (TDM, one issue per WG)
        if (jb == 0 && (t + 1 < SEQ)) {
            tdm_load_emit_tile(gbase + (size_t)(t + 1) * NTAGS,
                               ((t + 1) & 1) ? OFF_EMIT1 : OFF_EMIT0);
        }
        // (a) {rowmax2, mask} per row (float4 max-tree over published alpha)
        if (tid < ROWS) {
            const float* arow = alpha + tid * AS;
            float4 m4 = *(const float4*)(arow);
            #pragma unroll
            for (int i = 4; i < NTAGS; i += 4) {
                const float4 v = *(const float4*)(arow + i);
                m4.x = fmaxf(m4.x, v.x); m4.y = fmaxf(m4.y, v.y);
                m4.z = fmaxf(m4.z, v.z); m4.w = fmaxf(m4.w, v.w);
            }
            const float m  = fmaxf(fmaxf(m4.x, m4.y), fmaxf(m4.z, m4.w));
            const float mk = (float)mask[(size_t)(b0 + tid) * SEQ + t];
            rm_s[tid] = make_float2(m, mk);
        }
        // (a2) retire the tile for this timestep before publishing via barrier
        if (jb == 0) {
            if (t + 1 < SEQ) __builtin_amdgcn_s_wait_tensorcnt(1);
            else             __builtin_amdgcn_s_wait_tensorcnt(0);
        }
        __syncthreads();

        // (c) expA = 2^(alpha2 - rowmax2) -> f16 (float4 in, packed b64 out)
        for (int base = tid * 4; base < ROWS * NTAGS; base += THREADS * 4) {
            const int row = base >> 6, i = base & 63;
            const float4 a4 = *(const float4*)(alpha + row * AS + i);
            const float  m  = rm_s[row].x;
            v4h h;
            h[0] = (_Float16)__builtin_amdgcn_exp2f(a4.x - m);
            h[1] = (_Float16)__builtin_amdgcn_exp2f(a4.y - m);
            h[2] = (_Float16)__builtin_amdgcn_exp2f(a4.z - m);
            h[3] = (_Float16)__builtin_amdgcn_exp2f(a4.w - m);
            *(v4h*)(expA + row * EAS + i) = h;
        }
        __syncthreads();

        // (e) A fragments (two ds_load_b128 per K-chunk) and WMMA, K=64
        v16h a0, a1;
        {
            const _Float16* arow = expA + mrow * EAS;
            ((v8h*)&a0)[0] = *(const v8h*)(arow + kbase);
            ((v8h*)&a0)[1] = *(const v8h*)(arow + kbase + 8);
            ((v8h*)&a1)[0] = *(const v8h*)(arow + 32 + kbase);
            ((v8h*)&a1)[1] = *(const v8h*)(arow + 32 + kbase + 8);
        }
        v8f acc = {};
        acc = __builtin_amdgcn_wmma_f32_16x16x32_f16(false, a0, false, bfrag0,
                                                     (short)0, acc, false, false);
        acc = __builtin_amdgcn_wmma_f32_16x16x32_f16(false, a1, false, bfrag1,
                                                     (short)0, acc, false, false);

        // alpha2 update in registers; publish to LDS for next broadcast pass
        union { v8f v; float f[8]; } u; u.v = acc;
        #pragma unroll
        for (int r = 0; r < 8; ++r) {
            const int    M  = r + half8;
            const float2 rm = rm_s[M];
            const float  na = fmaf(emit_s[M * EMP + j], LOG2E,
                                   rm.x + __builtin_amdgcn_logf(u.f[r]));
            alpha_reg[r] = rm.y * na + (1.0f - rm.y) * alpha_reg[r];
            alpha[M * AS + j] = alpha_reg[r];
        }
        __syncthreads();
    }

    // log-denominator: ln2 * log2sumexp2(alpha2 + end*log2e) per row
    if (tid < ROWS) {
        float m = -3.4e38f;
        for (int i = 0; i < NTAGS; ++i)
            m = fmaxf(m, alpha[tid * AS + i] + end_t[i] * LOG2E);
        float s = 0.f;
        for (int i = 0; i < NTAGS; ++i)
            s += __builtin_amdgcn_exp2f(alpha[tid * AS + i] + end_t[i] * LOG2E - m);
        denom_out[b0 + tid] = LN2 * (m + __builtin_amdgcn_logf(s));
    }
}

// ---------------------------------------------------------------------------
// Joint (numerator) score: one block per batch row, strided gather + reduce.
// ---------------------------------------------------------------------------
__global__ __launch_bounds__(128) void crf_numer_kernel(
    const float*     __restrict__ inputs,
    const long long* __restrict__ tags,
    const int*       __restrict__ mask,
    const float*     __restrict__ transitions,
    const float*     __restrict__ start_t,
    const float*     __restrict__ end_t,
    float*           __restrict__ numer_out)
{
    __shared__ float sred[128];
    __shared__ int   mred[128];
    const int b = blockIdx.x, tid = threadIdx.x;
    const long long* trow = tags + (size_t)b * SEQ;
    const int*       mrow = mask + (size_t)b * SEQ;

    float part = 0.f; int msum = 0;
    for (int t = tid; t < SEQ; t += 128) {
        const int   tg = (int)trow[t];
        const int   mi = mrow[t];
        const float fm = (float)mi;
        msum += mi;
        if (t >= 1) {
            const int tp = (int)trow[t - 1];
            part += transitions[tp * NTAGS + tg] * fm;                 // trans * fmask[:,1:]
        }
        if (t <= SEQ - 2) {
            part += inputs[((size_t)b * SEQ + t) * NTAGS + tg] * fm;   // emit[:, :-1] * fmask[:, :-1]
        }
    }
    sred[tid] = part; mred[tid] = msum;
    __syncthreads();
    for (int s = 64; s > 0; s >>= 1) {
        if (tid < s) { sred[tid] += sred[tid + s]; mred[tid] += mred[tid + s]; }
        __syncthreads();
    }
    if (tid == 0) {
        float score = sred[0] + start_t[(int)trow[0]];
        const int last_idx = mred[0] - 1;
        const int last_tag = (int)trow[last_idx];
        score += end_t[last_tag]
               + inputs[((size_t)b * SEQ + (SEQ - 1)) * NTAGS + last_tag] * (float)mrow[SEQ - 1];
        numer_out[b] = score;
    }
}

// ---------------------------------------------------------------------------
// Final scalar: sum(numerator - denominator). Deterministic tree reduction.
// ---------------------------------------------------------------------------
__global__ __launch_bounds__(256) void crf_reduce_kernel(
    const float* __restrict__ numer,
    const float* __restrict__ denom,
    float*       __restrict__ out)
{
    __shared__ float sred[256];
    const int tid = threadIdx.x;
    sred[tid] = (numer[tid] - denom[tid]) + (numer[tid + 256] - denom[tid + 256]);
    __syncthreads();
    for (int k = 128; k > 0; k >>= 1) {
        if (tid < k) sred[tid] += sred[tid + k];
        __syncthreads();
    }
    if (tid == 0) out[0] = sred[0];
}

extern "C" void kernel_launch(void* const* d_in, const int* in_sizes, int n_in,
                              void* d_out, int out_size, void* d_ws, size_t ws_size,
                              hipStream_t stream)
{
    const float*     inputs      = (const float*)d_in[0];
    const long long* tags        = (const long long*)d_in[1];   // int64
    const int*       mask        = (const int*)d_in[2];         // int32
    const float*     transitions = (const float*)d_in[3];
    const float*     start_t     = (const float*)d_in[4];
    const float*     end_t       = (const float*)d_in[5];

    float* denom = (float*)d_ws;          // [BATCH]
    float* numer = denom + BATCH;         // [BATCH]

    crf_forward_kernel<<<BATCH / ROWS, THREADS, SMEM_BYTES, stream>>>(
        inputs, mask, transitions, start_t, end_t, denom);
    crf_numer_kernel<<<BATCH, 128, 0, stream>>>(
        inputs, tags, mask, transitions, start_t, end_t, numer);
    crf_reduce_kernel<<<1, 256, 0, stream>>>(numer, denom, (float*)d_out);
}
